// DotProductAttention_10969346474847
// MI455X (gfx1250) — matrix operand order repaired and checked by reading, not compiled, not run on hardware
//
#include <hip/hip_runtime.h>
#include <hip/hip_bf16.h>

typedef __bf16 bf16_t;
typedef __attribute__((ext_vector_type(16))) __bf16 v16bf;
typedef __attribute__((ext_vector_type(8)))  __bf16 v8bf;
typedef __attribute__((ext_vector_type(4)))  __bf16 v4bf;
typedef __attribute__((ext_vector_type(8)))  float  v8f;

#define SDIM 2048
#define DDIM 1024
#define JT   32          // j-tile (keys per iteration)
#define DSLICE 128       // D columns per wave (DDIM / 8 waves)

union V16U { v16bf v; v8bf h[2]; };

// A-operand fragment (16xK, 16-bit), row-major source with row stride rs.
// lanes 0-15: row=lane, k = k0+0..7 then k0+16..23; lanes 16-31: row=lane-16, k = k0+8..15 then k0+24..31.
__device__ __forceinline__ v16bf load_afrag(const bf16_t* base, int rs, int k0, int lane) {
  int row  = lane & 15;
  int koff = (lane & 16) ? 8 : 0;
  const bf16_t* p = base + row * rs + k0 + koff;
  V16U u;
  u.h[0] = *(const v8bf*)(p);
  u.h[1] = *(const v8bf*)(p + 16);
  return u.v;
}

// B-operand fragment (32x16, 16-bit): B(k,n) = base[(n0+n)*rs + k0 + k].
// lanes 0-15: col n=lane, k=0..7 then 16..23; lanes 16-31: col n=lane-16, k=8..15 then 24..31.
__device__ __forceinline__ v16bf load_bfrag(const bf16_t* base, int rs, int n0, int k0, int lane) {
  int n  = lane & 15;
  int kb = (lane & 16) ? 8 : 0;
  const bf16_t* p = base + (n0 + n) * rs + k0 + kb;
  V16U u;
  u.h[0] = *(const v8bf*)(p);
  u.h[1] = *(const v8bf*)(p + 16);
  return u.v;
}

__global__ void zero_f32_kernel(float* __restrict__ p, int n) {
  int i = blockIdx.x * blockDim.x + threadIdx.x;
  if (i < n) p[i] = 0.0f;
}

__global__ __launch_bounds__(256, 1)
void flash_attn_mean_kernel(const float* __restrict__ x, float* __restrict__ out) {
  extern __shared__ __align__(16) char smem[];
  // LDS overlay: Q staging aliases the first 32 KB of Krow (Q LDS is dead after
  // per-wave Q fragments are cached in VGPRs; loop-top barrier orders the reuse).
  bf16_t* Krow = (bf16_t*)smem;                      // 32*1024 bf16 = 64 KB (row-major K)
  bf16_t* Qs   = Krow;                               // 16*1024 bf16 = 32 KB (aliased)
  bf16_t* Vcol = Krow + JT * DDIM;                   // 1024*32 bf16 = 64 KB (col-major V: [d][j])
  float*  Sred = (float*)(Vcol + JT * DDIM);         // 16*32 f32    =  2 KB
  bf16_t* Pbuf = (bf16_t*)(Sred + 16 * JT);          // 16*32 bf16   =  1 KB
  // total 131 KB -> 2 workgroups per 320 KB WGP

  const int tid  = threadIdx.x;
  const int lane = tid & 31;
  const int wave = tid >> 5;
  const int b    = blockIdx.x >> 7;          // 128 row-tiles per batch
  const int i0   = (blockIdx.x & 127) << 4;  // first query row of this tile

  const float* Xb = x + (size_t)b * SDIM * DDIM;
  const float* Qg = Xb + (size_t)i0 * DDIM;

  // ---- stage Q tile (16x1024 fp32 -> bf16), coalesced, packed 8B LDS stores ----
  for (int e = tid * 4; e < 16 * DDIM; e += 256 * 4) {
    float4 f = *(const float4*)(Qg + e);
    v4bf q4 = { (bf16_t)f.x, (bf16_t)f.y, (bf16_t)f.z, (bf16_t)f.w };
    *(v4bf*)(Qs + e) = q4;
  }
  __syncthreads();

  const int dbase = wave * DSLICE;   // this wave's 128-wide D slice

  // cache Q A-fragments for the 4 k-chunks of this wave's D slice (32 VGPRs)
  v16bf qf[4];
#pragma unroll
  for (int c = 0; c < 4; ++c) qf[c] = load_afrag(Qs, DDIM, dbase + 32 * c, lane);

  // O accumulators: 8 tiles of 16x16 f32 (rows i, cols dbase+16t) = 64 VGPRs
  v8f o[8];
#pragma unroll
  for (int t = 0; t < 8; ++t) o[t] = (v8f){0.f,0.f,0.f,0.f,0.f,0.f,0.f,0.f};

  float m_run = -3.0e38f;   // per-row running max (row = lane & 15, mirrored upper half)
  float l_run = 0.0f;
  const int row = lane & 15;

  for (int j0 = 0; j0 < SDIM; j0 += JT) {
    __syncthreads();  // previous iteration's LDS consumers done (also guards Q alias)

    // ---- stage K/V tile (32x1024) as 4x4 subtiles: 4 coalesced b128 global loads,
    //      then packed 8B ds stores into both orientations ----
    const float* Kg = Xb + (size_t)j0 * DDIM;
    const int c0 = tid * 4;  // column quad owned by this thread (0..1020)
    for (int rg = 0; rg < 8; ++rg) {
      const int r0 = rg * 4;
      const float* gp = Kg + (size_t)r0 * DDIM + c0;
      float4 f0 = *(const float4*)(gp);
      float4 f1 = *(const float4*)(gp + DDIM);
      float4 f2 = *(const float4*)(gp + 2 * DDIM);
      float4 f3 = *(const float4*)(gp + 3 * DDIM);
      v4bf r0v = { (bf16_t)f0.x, (bf16_t)f0.y, (bf16_t)f0.z, (bf16_t)f0.w };
      v4bf r1v = { (bf16_t)f1.x, (bf16_t)f1.y, (bf16_t)f1.z, (bf16_t)f1.w };
      v4bf r2v = { (bf16_t)f2.x, (bf16_t)f2.y, (bf16_t)f2.z, (bf16_t)f2.w };
      v4bf r3v = { (bf16_t)f3.x, (bf16_t)f3.y, (bf16_t)f3.z, (bf16_t)f3.w };
      // row-major copy: 4 x ds_store_b64
      *(v4bf*)(Krow + (r0 + 0) * DDIM + c0) = r0v;
      *(v4bf*)(Krow + (r0 + 1) * DDIM + c0) = r1v;
      *(v4bf*)(Krow + (r0 + 2) * DDIM + c0) = r2v;
      *(v4bf*)(Krow + (r0 + 3) * DDIM + c0) = r3v;
      // column-major copy (transpose in registers): 4 x ds_store_b64
      v4bf vc0 = { r0v[0], r1v[0], r2v[0], r3v[0] };
      v4bf vc1 = { r0v[1], r1v[1], r2v[1], r3v[1] };
      v4bf vc2 = { r0v[2], r1v[2], r2v[2], r3v[2] };
      v4bf vc3 = { r0v[3], r1v[3], r2v[3], r3v[3] };
      *(v4bf*)(Vcol + (c0 + 0) * JT + r0) = vc0;
      *(v4bf*)(Vcol + (c0 + 1) * JT + r0) = vc1;
      *(v4bf*)(Vcol + (c0 + 2) * JT + r0) = vc2;
      *(v4bf*)(Vcol + (c0 + 3) * JT + r0) = vc3;
    }
    for (int e = tid; e < 16 * JT; e += 256) Sred[e] = 0.0f;

    // prefetch next tile while we compute (global_prefetch_b8)
    if (j0 + JT < SDIM) {
      const float* Kn = Kg + (size_t)JT * DDIM;
#pragma unroll
      for (int q = 0; q < 4; ++q)
        __builtin_prefetch(Kn + (((size_t)tid * 4 + q) << 5), 0, 0);
    }
    __syncthreads();

    // ---- partial scores S_w[16x32] over this wave's D slice (software-pipelined) ----
    v8f s0 = (v8f){0.f,0.f,0.f,0.f,0.f,0.f,0.f,0.f};
    v8f s1 = (v8f){0.f,0.f,0.f,0.f,0.f,0.f,0.f,0.f};
    v16bf kb0 = load_bfrag(Krow, DDIM, 0,  dbase, lane);
    v16bf kb1 = load_bfrag(Krow, DDIM, 16, dbase, lane);
#pragma unroll
    for (int c = 0; c < 4; ++c) {
      v16bf nb0 = kb0, nb1 = kb1;
      if (c < 3) {
        int d0 = dbase + 32 * (c + 1);
        nb0 = load_bfrag(Krow, DDIM, 0,  d0, lane);   // issued before the waits below
        nb1 = load_bfrag(Krow, DDIM, 16, d0, lane);
      }
      s0 = __builtin_amdgcn_wmma_f32_16x16x32_bf16(false, qf[c], false, kb0, (short)0, s0, false, false);
      s1 = __builtin_amdgcn_wmma_f32_16x16x32_bf16(false, qf[c], false, kb1, (short)0, s1, false, false);
      kb0 = nb0; kb1 = nb1;
    }

    // ---- cross-wave reduction of partial scores (ds_add_f32) ----
    {
      int m = (lane & 16) ? 8 : 0;  // C layout: VGPR r -> row r (lanes<16) / r+8 (lanes>=16)
      int n = lane & 15;
#pragma unroll
      for (int r = 0; r < 8; ++r) {
        atomicAdd(&Sred[(m + r) * JT + n],      s0[r]);
        atomicAdd(&Sred[(m + r) * JT + n + 16], s1[r]);
      }
    }
    __syncthreads();

    // ---- online softmax, one row per lane (upper half mirrors lower) ----
    float sv[JT];
#pragma unroll
    for (int jj = 0; jj < JT; ++jj) sv[jj] = Sred[row * JT + jj];

    float tmax = sv[0];
#pragma unroll
    for (int jj = 1; jj < JT; ++jj) tmax = fmaxf(tmax, sv[jj]);
    float newm  = fmaxf(m_run, tmax);
    float alpha = __expf(m_run - newm);
    float psum  = 0.0f;
    bf16_t pb[JT];
#pragma unroll
    for (int jj = 0; jj < JT; ++jj) {
      float e = __expf(sv[jj] - newm);
      psum += e;
      pb[jj] = (bf16_t)e;
    }
    m_run = newm;
    l_run = l_run * alpha + psum;

    if (wave == 0 && lane < 16) {
#pragma unroll
      for (int jj = 0; jj < JT; jj += 4) {
        v4bf p4 = { pb[jj], pb[jj + 1], pb[jj + 2], pb[jj + 3] };
        *(v4bf*)(Pbuf + row * JT + jj) = p4;
      }
    }
    __syncthreads();  // P ready for all waves

    // ---- rescale O by alpha (per-row broadcast via shfl) ----
#pragma unroll
    for (int r = 0; r < 8; ++r) {
      float ar = __shfl(alpha, r + ((lane & 16) >> 1), 32);  // row r or r+8
#pragma unroll
      for (int t = 0; t < 8; ++t) o[t][r] *= ar;
    }

    // ---- O += P (16x32) x V (32 x 128-slice), software-pipelined B loads ----
    v16bf pf = load_afrag(Pbuf, JT, 0, lane);
    v16bf vf = load_bfrag(Vcol, JT, dbase, 0, lane);
#pragma unroll
    for (int t = 0; t < 8; ++t) {
      v16bf vn = vf;
      if (t < 7) vn = load_bfrag(Vcol, JT, dbase + (t + 1) * 16, 0, lane);
      o[t] = __builtin_amdgcn_wmma_f32_16x16x32_bf16(false, pf, false, vf, (short)0, o[t], false, false);
      vf = vn;
    }
  }

  // ---- epilogue: (sum over 16 rows of context/l) / S, atomic into out ----
  float linv = 1.0f / l_run;  // per-row (row = lane & 15)
#pragma unroll
  for (int t = 0; t < 8; ++t) {
    float csum = 0.0f;
#pragma unroll
    for (int r = 0; r < 8; ++r) {
      float lr = __shfl(linv, r + ((lane & 16) >> 1), 32);
      csum += o[t][r] * lr;
    }
    csum += __shfl_xor(csum, 16, 32);  // combine row-halves (same column in both lane halves)
    if (lane < 16) {
      unsafeAtomicAdd(&out[b * DDIM + dbase + t * 16 + (lane & 15)],
                      csum * (1.0f / (float)SDIM));
    }
  }
}

extern "C" void kernel_launch(void* const* d_in, const int* in_sizes, int n_in,
                              void* d_out, int out_size, void* d_ws, size_t ws_size,
                              hipStream_t stream) {
  const float* x = (const float*)d_in[0];
  float* out = (float*)d_out;

  // d_out is poisoned by the harness; we accumulate with atomics, so zero it first.
  zero_f32_kernel<<<(out_size + 255) / 256, 256, 0, stream>>>(out, out_size);

  const int batches  = in_sizes[0] / (SDIM * DDIM);  // 16
  const int n_blocks = batches * (SDIM / 16);        // 2048 workgroups
  const size_t smem =
      (size_t)(JT * DDIM) * sizeof(bf16_t) +   // Krow 64 KB (Q staging aliased inside)
      (size_t)(JT * DDIM) * sizeof(bf16_t) +   // Vcol 64 KB
      (size_t)(16 * JT)   * sizeof(float)  +   // Sred  2 KB
      (size_t)(16 * JT)   * sizeof(bf16_t);    // Pbuf  1 KB

  flash_attn_mean_kernel<<<n_blocks, 256, smem, stream>>>(x, out);
}
